// LinearRNN_2302102470925
// MI455X (gfx1250) — matrix operand order repaired and checked by reading, not compiled, not run on hardware
//
#include <hip/hip_runtime.h>

// ---------------------------------------------------------------------------
// Linear RNN on MI455X (gfx1250, wave32, WMMA + Tensor Data Mover).
// Chunk-parallel linear-scan decomposition; all matmuls via
// v_wmma_f32_16x16x32_bf16 (bf16 inputs, f32 accumulate). TDM stages
// streaming operands into LDS (tensorcnt path), overlapped with compute.
// A-operands are hoisted into registers so LDS loads clause up ahead of the
// WMMA chains; per-step state exchange is double-buffered (1 barrier/step);
// global stores are issued after the barrier so only dscnt is drained on the
// sequential critical path.
// ---------------------------------------------------------------------------

typedef __bf16 bf16;
typedef __attribute__((ext_vector_type(16))) __bf16 v16bf;
typedef __attribute__((ext_vector_type(8)))  float  v8f;

#define BB    64            // batch
#define TT    4096          // time steps
#define HH    128           // hidden = input = output width
#define CHUNK 64            // time-chunk length
#define PCH   (TT / CHUNK)  // number of chunks = 64
#define SUB   4             // timesteps staged per TDM transfer
#define LDST  136           // LDS row stride (elements): 272B rows -> 16B aligned + bank rotation

// ---- TDM availability ------------------------------------------------------
#if defined(__has_builtin)
#if __has_builtin(__builtin_amdgcn_tensor_load_to_lds) && __has_builtin(__builtin_amdgcn_s_wait_tensorcnt)
#define HAVE_TDM 1
#endif
#endif
#ifndef HAVE_TDM
#define HAVE_TDM 0
#endif

#if HAVE_TDM
typedef unsigned int u32x4 __attribute__((ext_vector_type(4)));
typedef int          i32x4 __attribute__((ext_vector_type(4)));
typedef int          i32x8 __attribute__((ext_vector_type(8)));

// Build a <=2D D# (groups 0+1) per CDNA5 ISA 8.3/8.4 and issue TENSOR_LOAD_TO_LDS.
__device__ __forceinline__ void tdm_load_2d(unsigned lds_addr, const void* gaddr,
                                            unsigned g1w0, unsigned tdim0, unsigned tdim1,
                                            unsigned tile0, unsigned tile1,
                                            unsigned long stride0) {
  const unsigned long ga = (unsigned long)gaddr;
  u32x4 g0;
  g0[0] = 1u;                                                  // count=1, user D#
  g0[1] = lds_addr;                                            // lds_addr[31:0]
  g0[2] = (unsigned)(ga & 0xFFFFFFFFu);                        // global_addr[31:0]
  g0[3] = (unsigned)((ga >> 32) & 0x01FFFFFFu) | 0x80000000u;  // global_addr[56:32] | type=2
  i32x8 g1;
  g1[0] = (int)g1w0;
  g1[1] = (int)((tdim0 & 0xFFFFu) << 16);                      // atomic_barrier_addr=0 | tensor_dim0 lo
  g1[2] = (int)(((tdim0 >> 16) & 0xFFFFu) | ((tdim1 & 0xFFFFu) << 16));
  g1[3] = (int)(((tdim1 >> 16) & 0xFFFFu) | ((tile0 & 0xFFFFu) << 16));
  g1[4] = (int)(tile1 & 0xFFFFu);                              // tile_dim1 | tile_dim2=0
  g1[5] = (int)(unsigned)(stride0 & 0xFFFFFFFFu);
  g1[6] = (int)(unsigned)((stride0 >> 32) & 0xFFFFu);          // stride0 hi | stride1=0
  g1[7] = 0;
  i32x4 z4 = {0, 0, 0, 0};
#if __clang_major__ >= 23
  i32x8 z8 = {0, 0, 0, 0, 0, 0, 0, 0};
  __builtin_amdgcn_tensor_load_to_lds(g0, g1, z4, z4, z8, 0);
#else
  __builtin_amdgcn_tensor_load_to_lds(g0, g1, z4, z4, 0);
#endif
}
__device__ __forceinline__ unsigned lds_off(const void* p) {
  return (unsigned)(unsigned long)p;  // LDS aperture: low 32 bits == LDS byte offset
}
#endif  // HAVE_TDM

__device__ __forceinline__ v8f wmma_bf(v16bf a, v16bf b, v8f c) {
  // (neg_a, A, neg_b, B, c_mod, C, reuse_a, reuse_b)
  return __builtin_amdgcn_wmma_f32_16x16x32_bf16(false, a, false, b, (short)0, c, false, false);
}

__device__ __forceinline__ v8f splat8(float v) {
  v8f c;
#pragma unroll
  for (int j = 0; j < 8; ++j) c[j] = v;
  return c;
}

// ---- operand loaders (CDNA5 16-bit WMMA register layouts) ------------------
// A (16x32, MxK): rows M=lane&15; VGPR0-3: K=half*8+0..7, VGPR4-7: K=16+half*8+0..7
__device__ __forceinline__ v16bf opA_lds(const bf16* m, int rowbase, int kc, int lane) {
  const bf16* p = m + (rowbase + (lane & 15)) * LDST + kc * 32 + ((lane >> 4) & 1) * 8;
  v16bf o;
  reinterpret_cast<uint4*>(&o)[0] = *reinterpret_cast<const uint4*>(p);
  reinterpret_cast<uint4*>(&o)[1] = *reinterpret_cast<const uint4*>(p + 16);
  return o;
}
__device__ __forceinline__ v16bf opA_gbf(const bf16* m, long rowstride, long rowbase, int kc, int lane) {
  const bf16* p = m + (rowbase + (lane & 15)) * rowstride + kc * 32 + ((lane >> 4) & 1) * 8;
  v16bf o;
  reinterpret_cast<uint4*>(&o)[0] = *reinterpret_cast<const uint4*>(p);
  reinterpret_cast<uint4*>(&o)[1] = *reinterpret_cast<const uint4*>(p + 16);
  return o;
}
__device__ __forceinline__ v16bf opA_gf32(const float* m, long rowstride, long rowbase, int kc, int lane) {
  const float* p = m + (rowbase + (lane & 15)) * rowstride + kc * 32 + ((lane >> 4) & 1) * 8;
  v16bf o;
#pragma unroll
  for (int i = 0; i < 8; ++i) { o[i] = (bf16)p[i]; o[8 + i] = (bf16)p[16 + i]; }
  return o;
}
// B (32x16, KxN): lane holds column N=lane&15; VGPR0-3: K=half*8+0..7, VGPR4-7: K=16+half*8+0..7
// Sourced from row-major W where B = W^T (column n of B == row n of W).
__device__ __forceinline__ v16bf opB_lds(const bf16* m, int colbase, int kc, int lane) {
  const bf16* p = m + (colbase + (lane & 15)) * LDST + kc * 32 + ((lane >> 4) & 1) * 8;
  v16bf o;
  reinterpret_cast<uint4*>(&o)[0] = *reinterpret_cast<const uint4*>(p);
  reinterpret_cast<uint4*>(&o)[1] = *reinterpret_cast<const uint4*>(p + 16);
  return o;
}
__device__ __forceinline__ v16bf opB_g(const bf16* W, int colbase, int kc, int lane) {
  const bf16* p = W + (colbase + (lane & 15)) * HH + kc * 32 + ((lane >> 4) & 1) * 8;
  v16bf o;
  reinterpret_cast<uint4*>(&o)[0] = *reinterpret_cast<const uint4*>(p);
  reinterpret_cast<uint4*>(&o)[1] = *reinterpret_cast<const uint4*>(p + 16);
  return o;
}

// ---- workspace layout ------------------------------------------------------
constexpr size_t OFF_WI   = 0;
constexpr size_t OFF_WH   = OFF_WI + (size_t)HH * HH * sizeof(bf16);
constexpr size_t OFF_WO   = OFF_WH + (size_t)HH * HH * sizeof(bf16);
constexpr size_t OFF_BU   = OFF_WO + (size_t)HH * HH * sizeof(bf16);
constexpr size_t OFF_WPOW = OFF_BU + 1024;
constexpr size_t OFF_F    = OFF_WPOW + (size_t)CHUNK * HH * HH * sizeof(bf16);
constexpr size_t OFF_S    = OFF_F    + (size_t)PCH * BB * HH * sizeof(float);
constexpr size_t OFF_HS   = OFF_S    + (size_t)(PCH + 1) * BB * HH * sizeof(float);
// total ~73.6 MB

// ---- kernel 0: weight conversion + fused bias ------------------------------
__global__ void k_prep(const float* __restrict__ Wi, const float* __restrict__ bi,
                       const float* __restrict__ Wh, const float* __restrict__ bh,
                       const float* __restrict__ Wo,
                       bf16* __restrict__ Wi_bf, bf16* __restrict__ Wh_bf,
                       bf16* __restrict__ Wo_bf, float* __restrict__ biasU) {
  int i = blockIdx.x * 256 + threadIdx.x;
  if (i < HH * HH) {
    Wi_bf[i] = (bf16)Wi[i];
    Wh_bf[i] = (bf16)Wh[i];
    Wo_bf[i] = (bf16)Wo[i];
  }
  if (i < HH) biasU[i] = bi[i] + bh[i];
}

// ---- kernel 1: fused input-projection + per-chunk local scan ---------------
// grid (PCH, BB/16); 256 threads = 8 waves; wave w owns N-tile w.
// x staged per SUB timesteps by the TDM; converted once into a bank-padded
// bf16 A-buffer (the convert pass double-buffers the DMA). Per-step h tile is
// double-buffered in LDS -> single barrier per timestep; HS global stores are
// issued after the barrier so the barrier only drains dscnt.
__global__ __launch_bounds__(256) void k_scan_local(
    const float* __restrict__ x, const bf16* __restrict__ Wi_bf,
    const bf16* __restrict__ Wh_bf, const float* __restrict__ biasU,
    bf16* __restrict__ HS, float* __restrict__ F) {
#if HAVE_TDM
  __shared__ __align__(16) float xstage[16 * SUB * HH];   // 32 KB raw f32 tile
#endif
  __shared__ __align__(16) bf16 xbf[SUB][16 * LDST];      // 17.4 KB bf16 A tiles
  __shared__ __align__(16) bf16 sh[2][16 * LDST];         // double-buffered h tile
  const int tid  = threadIdx.x;
  const int lane = tid & 31;
  const int w    = tid >> 5;
  const int p    = blockIdx.x;
  const int b0   = blockIdx.y * 16;

  v16bf bWi[4], bWh[4];                                   // weight B-operands in regs
#pragma unroll
  for (int k = 0; k < 4; ++k) {
    bWi[k] = opB_g(Wi_bf, w * 16, k, lane);
    bWh[k] = opB_g(Wh_bf, w * 16, k, lane);
  }
  const float bias = biasU[w * 16 + (lane & 15)];
  const int   n    = w * 16 + (lane & 15);
  const int   mh   = ((lane >> 4) & 1) * 8;
  const long  t0   = (long)p * CHUNK;

#if HAVE_TDM
  const unsigned xs_off = lds_off(xstage);
  if (w == 0)  // first sub-chunk: 2D tile, 16 rows x SUB*128 f32 (data_size=4B)
    tdm_load_2d(xs_off, x + ((long)b0 * TT + t0) * HH, 0x00020000u,
                (unsigned)((long)TT * HH), 16u, SUB * HH, 16u, (unsigned long)TT * HH);
#endif

  for (int sc = 0; sc < CHUNK / SUB; ++sc) {
#if HAVE_TDM
    if (w == 0) __builtin_amdgcn_s_wait_tensorcnt(0);
#endif
    __syncthreads();
    // one-shot f32 -> bf16 convert into padded A layout (each element once)
    for (int i = tid; i < 16 * SUB * HH; i += 256) {
      const int row = i >> 9;          // SUB*HH == 512
      const int rem = i & 511;
#if HAVE_TDM
      const float v = xstage[i];
#else
      const float v = x[((long)(b0 + row) * TT + t0 + sc * SUB) * HH + rem];
#endif
      xbf[rem >> 7][row * LDST + (rem & 127)] = (bf16)v;
    }
    __syncthreads();
#if HAVE_TDM
    if (w == 0 && sc + 1 < CHUNK / SUB)  // overlap next DMA with compute
      tdm_load_2d(xs_off, x + ((long)b0 * TT + t0 + (sc + 1) * SUB) * HH, 0x00020000u,
                  (unsigned)((long)TT * HH), 16u, SUB * HH, 16u, (unsigned long)TT * HH);
#endif
#pragma unroll 1
    for (int so = 0; so < SUB; ++so) {
      const int  s   = sc * SUB + so;
      const long t   = t0 + s;
      const bool rec = (s > 0);
      // hoist all A-operands: LDS loads clause up, partial dscnt waits, then
      // the 8 WMMAs issue back-to-back
      v16bf ax[4], ah[4];
#pragma unroll
      for (int k = 0; k < 4; ++k) ax[k] = opA_lds(xbf[so], 0, k, lane);
      if (rec) {
#pragma unroll
        for (int k = 0; k < 4; ++k) ah[k] = opA_lds(sh[s & 1], 0, k, lane);
      }
      v8f c = splat8(bias);
#pragma unroll
      for (int k = 0; k < 4; ++k) c = wmma_bf(ax[k], bWi[k], c);   // u_t
      if (rec) {
#pragma unroll
        for (int k = 0; k < 4; ++k) c = wmma_bf(ah[k], bWh[k], c); // + Wh h_{t-1}
      }
      bf16* shw = sh[(s + 1) & 1];     // publish h_t into the other buffer (LDS only)
      bf16  hv[8];
#pragma unroll
      for (int j = 0; j < 8; ++j) {
        hv[j] = (bf16)c[j];
        shw[(mh + j) * LDST + n] = hv[j];
      }
      __syncthreads();                 // drains dscnt only; orders WAR/RAW on sh
      // global stores after the barrier: retire during the next timestep
#pragma unroll
      for (int j = 0; j < 8; ++j)
        HS[((long)(b0 + mh + j) * TT + t) * HH + n] = hv[j];
      if (s == CHUNK - 1) {            // chunk-final local state (f32)
#pragma unroll
        for (int j = 0; j < 8; ++j)
          F[((long)p * BB + b0 + mh + j) * HH + n] = c[j];
      }
    }
  }
}

// ---- kernel 2: Wpow[i] = Wh^(i+1), row-major bf16 --------------------------
__global__ __launch_bounds__(256) void k_powers(const bf16* __restrict__ Wh_bf,
                                                bf16* __restrict__ Wpow) {
  __shared__ __align__(16) bf16 sPT[128 * LDST];   // (Wh^i)^T, row-major
  const int tid = threadIdx.x, lane = tid & 31, w = tid >> 5;
  for (int i = tid; i < HH * HH; i += 256) {
    const int  r = i >> 7, cc = i & 127;
    const bf16 v = Wh_bf[i];
    sPT[cc * LDST + r] = v;
    Wpow[i] = v;                                   // Wpow[0] = Wh^1
  }
  v16bf aW[4];                                     // rows of Wh for M-tile w
#pragma unroll
  for (int k = 0; k < 4; ++k) aW[k] = opA_gbf(Wh_bf, HH, w * 16, k, lane);
  __syncthreads();
  for (int i = 1; i < CHUNK; ++i) {                // R = Wh * Wh^i
    v8f acc[8];
#pragma unroll
    for (int nt = 0; nt < 8; ++nt) {
      v16bf bb[4];
#pragma unroll
      for (int k = 0; k < 4; ++k) bb[k] = opB_lds(sPT, nt * 16, k, lane);
      v8f c = splat8(0.0f);
#pragma unroll
      for (int k = 0; k < 4; ++k) c = wmma_bf(aW[k], bb[k], c);
      acc[nt] = c;
    }
    __syncthreads();                               // all reads of sPT done
#pragma unroll
    for (int nt = 0; nt < 8; ++nt) {               // LDS transpose update first
#pragma unroll
      for (int j = 0; j < 8; ++j) {
        const int m  = w * 16 + ((lane >> 4) & 1) * 8 + j;
        const int nn = nt * 16 + (lane & 15);
        sPT[nn * LDST + m] = (bf16)acc[nt][j];
      }
    }
    __syncthreads();                               // publish sPT (dscnt only)
#pragma unroll
    for (int nt = 0; nt < 8; ++nt) {               // global stores off the barrier path
#pragma unroll
      for (int j = 0; j < 8; ++j) {
        const int m  = w * 16 + ((lane >> 4) & 1) * 8 + j;
        const int nn = nt * 16 + (lane & 15);
        Wpow[(long)i * HH * HH + m * HH + nn] = (bf16)acc[nt][j];
      }
    }
  }
}

// ---- kernel 3: chunk-boundary state scan  s_p = f_{p-1} + Wh^C * s_{p-1} ---
// Double-buffered state tile: one barrier per chunk step; S global stores
// issued after the barrier.
__global__ __launch_bounds__(256) void k_chunkscan(
    const bf16* __restrict__ WhC, const float* __restrict__ hidden,
    const float* __restrict__ F, float* __restrict__ S) {
  __shared__ __align__(16) bf16 ss[2][64 * LDST];
  const int tid = threadIdx.x, lane = tid & 31, w = tid >> 5;
  for (int i = tid; i < BB * HH; i += 256) {
    const float hv = hidden[i];                    // exact for any initial state
    S[i] = hv;
    ss[0][(i >> 7) * LDST + (i & 127)] = (bf16)hv;
  }
  v16bf bW[4];
#pragma unroll
  for (int k = 0; k < 4; ++k) bW[k] = opB_g(WhC, w * 16, k, lane);
  const int n  = w * 16 + (lane & 15);
  const int mh = ((lane >> 4) & 1) * 8;
  __syncthreads();
  for (int p = 1; p <= PCH; ++p) {
    const bf16* sr = ss[(p - 1) & 1];
    v8f acc[4];
#pragma unroll
    for (int mt = 0; mt < 4; ++mt) {
      v16bf as[4];
#pragma unroll
      for (int k = 0; k < 4; ++k) as[k] = opA_lds(sr, mt * 16, k, lane);
      v8f c;
#pragma unroll
      for (int j = 0; j < 8; ++j)
        c[j] = F[((long)(p - 1) * BB + mt * 16 + mh + j) * HH + n];
#pragma unroll
      for (int k = 0; k < 4; ++k) c = wmma_bf(as[k], bW[k], c);
      acc[mt] = c;
    }
    bf16* sw = ss[p & 1];
#pragma unroll
    for (int mt = 0; mt < 4; ++mt) {               // LDS publish before barrier
#pragma unroll
      for (int j = 0; j < 8; ++j)
        sw[(mt * 16 + mh + j) * LDST + n] = (bf16)acc[mt][j];
    }
    __syncthreads();
#pragma unroll
    for (int mt = 0; mt < 4; ++mt) {               // global stores after barrier
#pragma unroll
      for (int j = 0; j < 8; ++j)
        S[((long)p * BB + mt * 16 + mh + j) * HH + n] = acc[mt][j];  // S[PCH]==h_final
    }
  }
}

// ---- kernel 4: correction  hs[p,delta] += s_p * (Wh^(delta+1))^T -----------
// grid (CHUNK, PCH); LDS-free: weight tiles hoisted per wave.
__global__ __launch_bounds__(256) void k_correct(
    const bf16* __restrict__ Wpow, const float* __restrict__ S,
    bf16* __restrict__ HS) {
  const int tid = threadIdx.x, lane = tid & 31, w = tid >> 5;
  const int delta = blockIdx.x;
  const int p     = blockIdx.y;
  const bf16*  Wp = Wpow + (long)delta * HH * HH;
  const float* Sp = S + (long)p * BB * HH;
  const long   t  = (long)p * CHUNK + delta;
  v16bf bW[4];
#pragma unroll
  for (int k = 0; k < 4; ++k) bW[k] = opB_g(Wp, w * 16, k, lane);
  const int n  = w * 16 + (lane & 15);
  const int mh = ((lane >> 4) & 1) * 8;
#pragma unroll
  for (int mt = 0; mt < 4; ++mt) {
    v16bf as[4];
#pragma unroll
    for (int k = 0; k < 4; ++k) as[k] = opA_gf32(Sp, HH, mt * 16, k, lane);
    v8f c;
#pragma unroll
    for (int j = 0; j < 8; ++j)
      c[j] = (float)HS[((long)(mt * 16 + mh + j) * TT + t) * HH + n];
#pragma unroll
    for (int k = 0; k < 4; ++k) c = wmma_bf(as[k], bW[k], c);
#pragma unroll
    for (int j = 0; j < 8; ++j)
      HS[((long)(mt * 16 + mh + j) * TT + t) * HH + n] = (bf16)c[j];
  }
}

// ---- kernel 5: output GEMM  out = hs * Wo^T + bo ---------------------------
// 128 flattened (b*T+t) rows per WG. The 32 KB contiguous HS block is staged
// into LDS by the TDM as a 1D tile with pad_interval=64dw/pad_amount=4dw so it
// lands directly in the 272B-stride bank-rotated A layout. Wo stays L2-hot.
// out is written non-temporally (write-once, keep HS resident in L2).
__global__ __launch_bounds__(256) void k_out(
    const bf16* __restrict__ HS, const bf16* __restrict__ Wo_bf,
    const float* __restrict__ bo, float* __restrict__ out) {
  __shared__ __align__(16) bf16 sA[128 * LDST];
  const int tid = threadIdx.x, lane = tid & 31, w = tid >> 5;
  const long row0_blk = (long)blockIdx.x * 128;
#if HAVE_TDM
  if (w == 0) {
    // 1D 16384-element bf16 tile (data_size=2B), pad 4 dwords every 64 dwords
    tdm_load_2d(lds_off(sA), HS + row0_blk * HH, 0x07510000u,
                16384u, 0u, 16384u, 0u, 16384ul);
    __builtin_amdgcn_s_wait_tensorcnt(0);
  }
#else
  for (int i = tid; i < 128 * HH; i += 256)
    sA[(i >> 7) * LDST + (i & 127)] = HS[row0_blk * HH + i];
#endif
  __syncthreads();
  v16bf a[4];
#pragma unroll
  for (int k = 0; k < 4; ++k) a[k] = opA_lds(sA, w * 16, k, lane);
  const int mh = ((lane >> 4) & 1) * 8;
#pragma unroll
  for (int nt = 0; nt < 8; ++nt) {
    const int n = nt * 16 + (lane & 15);
    v16bf bb[4];
#pragma unroll
    for (int k = 0; k < 4; ++k) bb[k] = opB_g(Wo_bf, nt * 16, k, lane);
    v8f c = splat8(bo[n]);
#pragma unroll
    for (int k = 0; k < 4; ++k) c = wmma_bf(a[k], bb[k], c);
#pragma unroll
    for (int j = 0; j < 8; ++j)
      __builtin_nontemporal_store(c[j], &out[(row0_blk + w * 16 + mh + j) * HH + n]);
  }
}

// ---- kernel 6: h_final = S[PCH] --------------------------------------------
__global__ void k_hfinal(const float* __restrict__ S, float* __restrict__ out_tail) {
  int i = blockIdx.x * 256 + threadIdx.x;
  if (i < BB * HH)
    __builtin_nontemporal_store(S[(long)PCH * BB * HH + i], &out_tail[i]);
}

// ---------------------------------------------------------------------------
extern "C" void kernel_launch(void* const* d_in, const int* in_sizes, int n_in,
                              void* d_out, int out_size, void* d_ws, size_t ws_size,
                              hipStream_t stream) {
  (void)in_sizes; (void)n_in; (void)out_size; (void)ws_size;
  const float* x      = (const float*)d_in[0];
  const float* hidden = (const float*)d_in[1];
  const float* Wi     = (const float*)d_in[2];
  const float* bi     = (const float*)d_in[3];
  const float* Wh     = (const float*)d_in[4];
  const float* bh     = (const float*)d_in[5];
  const float* Wo     = (const float*)d_in[6];
  const float* bo     = (const float*)d_in[7];

  char*  ws    = (char*)d_ws;
  bf16*  Wi_bf = (bf16*)(ws + OFF_WI);
  bf16*  Wh_bf = (bf16*)(ws + OFF_WH);
  bf16*  Wo_bf = (bf16*)(ws + OFF_WO);
  float* biasU = (float*)(ws + OFF_BU);
  bf16*  Wpow  = (bf16*)(ws + OFF_WPOW);
  float* F     = (float*)(ws + OFF_F);
  float* S     = (float*)(ws + OFF_S);
  bf16*  HS    = (bf16*)(ws + OFF_HS);
  float* out   = (float*)d_out;

  k_prep<<<64, 256, 0, stream>>>(Wi, bi, Wh, bh, Wo, Wi_bf, Wh_bf, Wo_bf, biasU);
  k_scan_local<<<dim3(PCH, BB / 16), 256, 0, stream>>>(x, Wi_bf, Wh_bf, biasU, HS, F);
  k_powers<<<1, 256, 0, stream>>>(Wh_bf, Wpow);
  k_chunkscan<<<1, 256, 0, stream>>>(Wpow + (long)(CHUNK - 1) * HH * HH, hidden, F, S);
  k_correct<<<dim3(CHUNK, PCH), 256, 0, stream>>>(Wpow, S, HS);
  k_out<<<(BB * TT) / 128, 256, 0, stream>>>(HS, Wo_bf, bo, out);
  k_hfinal<<<(BB * HH + 255) / 256, 256, 0, stream>>>(S, out + (size_t)BB * TT * HH);
}
